// LinearCategoricalEncoding_12532714570307
// MI455X (gfx1250) — matrix operand order repaired and checked by reading, not compiled, not run on hardware
//
#include <hip/hip_runtime.h>
#include <hip/hip_bf16.h>
#include <math.h>

// ---------------------------------------------------------------------------
// Problem constants
// ---------------------------------------------------------------------------
#define BB 64
#define SS 2048
#define DD 64
#define EE 64
#define HH 512
#define FF 4
#define NTOK (BB * SS)   // 131072 tokens

typedef _Float16 half_t;
typedef __attribute__((ext_vector_type(16))) _Float16 v16h;
typedef __attribute__((ext_vector_type(8)))  _Float16 v8h;
typedef __attribute__((ext_vector_type(8)))  float    v8f;

union V16H { v16h v; v8h h[2]; };

// B fragment: 16 contiguous halves per lane (N-major packed weights)
static __device__ inline v16h ldB(const half_t* p) {
    V16H u;
    u.h[0] = *(const v8h*)p;
    u.h[1] = *(const v8h*)(p + 8);
    return u.v;
}
// A fragment: 16 contiguous halves at p (row-major source)
static __device__ inline v16h ldA(const half_t* p) {
    V16H u;
    u.h[0] = *(const v8h*)p;
    u.h[1] = *(const v8h*)(p + 8);
    return u.v;
}
static __device__ inline v8f zero8() {
    v8f z;
#pragma unroll
    for (int i = 0; i < 8; ++i) z[i] = 0.0f;
    return z;
}
static __device__ inline v8f wmma16(v16h a, v16h b, v8f c) {
    return __builtin_amdgcn_wmma_f32_16x16x32_f16(false, a, false, b, (short)0, c,
                                                  false, false);
}
static __device__ inline float gelu_exact(float x) {
    return 0.5f * x * (1.0f + erff(x * 0.70710678118654752f));
}
// CDNA5 async global->LDS copy, 16 bytes per active lane (ASYNCcnt tracked)
static __device__ inline void async_copy_b128(uint32_t lds_byte_addr,
                                              const void* gptr) {
    asm volatile("global_load_async_to_lds_b128 %0, %1, off"
                 :
                 : "v"(lds_byte_addr), "v"((uint64_t)(uintptr_t)gptr)
                 : "memory");
}
static __device__ inline void wait_async0() {
    asm volatile("s_wait_asynccnt 0" ::: "memory");
}

// ---------------------------------------------------------------------------
// Pre-pass: pack weight matrix (K x N, row-major f32) into B-operand layout:
// dst[n*K + k] = (f16) src[k*N + n]  (N-major, K contiguous per column)
// ---------------------------------------------------------------------------
__global__ void pack_bt(const float* __restrict__ src, half_t* __restrict__ dst,
                        int K, int N) {
    int i = blockIdx.x * blockDim.x + threadIdx.x;
    if (i >= K * N) return;
    int n = i / K;
    int k = i - n * K;
    dst[i] = (half_t)src[(size_t)k * N + n];
}

// Gather ext = embed[categ] as f16, row-major [NTOK][64]
__global__ void gather_ext(const float* __restrict__ embed,
                           const int* __restrict__ categ,
                           half_t* __restrict__ extq) {
    int i = blockIdx.x * blockDim.x + threadIdx.x;
    if (i >= NTOK * EE) return;
    int t = i >> 6, e = i & 63;
    extq[i] = (half_t)embed[(size_t)categ[t] * EE + e];
}

// ---------------------------------------------------------------------------
// LU (partial pivoting) on the four 64x64 conv matrices; initialize
// ldj[b] = S * sum_f log|det conv_W[f]| for every batch. Single workgroup.
// ---------------------------------------------------------------------------
__global__ void logdet_init(const float* __restrict__ convW,
                            float* __restrict__ ldj) {
    __shared__ float A[DD][DD + 1];
    __shared__ float logs;
    __shared__ int piv;
    int t = threadIdx.x;  // 64 threads
    if (t == 0) logs = 0.0f;
    __syncthreads();
    for (int f = 0; f < FF; ++f) {
        for (int j = 0; j < DD; ++j) A[j][t] = convW[f * DD * DD + j * DD + t];
        __syncthreads();
        for (int k = 0; k < DD; ++k) {
            if (t == 0) {
                int p = k;
                float mv = fabsf(A[k][k]);
                for (int i = k + 1; i < DD; ++i) {
                    float v = fabsf(A[i][k]);
                    if (v > mv) { mv = v; p = i; }
                }
                piv = p;
            }
            __syncthreads();
            if (piv != k) {
                float tmp = A[k][t];
                A[k][t] = A[piv][t];
                A[piv][t] = tmp;
            }
            __syncthreads();
            if (t == 0) logs += logf(fabsf(A[k][k]));
            if (t > k) {
                float fct = A[t][k] / A[k][k];
                for (int j = k; j < DD; ++j) A[t][j] -= fct * A[k][j];
            }
            __syncthreads();
        }
        __syncthreads();
    }
    float base = (float)SS * logs;
    if (t < BB) ldj[t] = base;
}

// ---------------------------------------------------------------------------
// Stage 1 (per flow): actnorm (ext @ W_an) -> tanh/exp z-update + ldj,
// then z <- z @ conv_W.  One wave per 16-token tile, 8 waves per block.
// ---------------------------------------------------------------------------
__global__ __launch_bounds__(256) void flow_stage1(
    const half_t* __restrict__ extq,    // [NTOK][64] f16
    float* __restrict__ zbuf,           // [NTOK][64] f32
    const half_t* __restrict__ anWt,    // [128][64] packed
    const float* __restrict__ anb,      // [128]
    const half_t* __restrict__ convWt,  // [64][64] packed
    float* __restrict__ ldj)            // [B]
{
    __shared__ __align__(16) half_t zt[8][16 * 72];

    int lane = threadIdx.x & 31;
    int w = threadIdx.x >> 5;
    int tile = blockIdx.x * 8 + w;
    int token0 = tile * 16;
    int b = token0 / SS;
    int mrow = lane & 15;
    int hi = (lane >> 4) & 1;
    int mr = hi * 8;

    // A fragments of ext (16x64, two K=32 steps)
    v16h aext[2];
#pragma unroll
    for (int kt = 0; kt < 2; ++kt)
        aext[kt] = ldA(extq + (size_t)(token0 + mrow) * EE + kt * 32 + hi * 16);

    // nn = ext @ actnorm_W : 16x128 output (8 column tiles)
    v8f acc[8];
#pragma unroll
    for (int nt = 0; nt < 8; ++nt) {
        v8f a = zero8();
#pragma unroll
        for (int kt = 0; kt < 2; ++kt) {
            const half_t* bp = anWt + (size_t)(nt * 16 + mrow) * 64 + kt * 32 + hi * 16;
            a = wmma16(aext[kt], ldB(bp), a);
        }
        acc[nt] = a;
    }

    // z = (z + bias) * exp(tanh(scales)); accumulate ldj; stash z as f16
    float ldjp = 0.0f;
#pragma unroll
    for (int nt = 0; nt < 4; ++nt) {
        int d = nt * 16 + mrow;
        float bb_ = anb[d];
        float bs_ = anb[64 + d];
#pragma unroll
        for (int r = 0; r < 8; ++r) {
            int m = mr + r;
            float zo = zbuf[(size_t)(token0 + m) * DD + d];
            float sc = tanhf(acc[nt + 4][r] + bs_);
            float zn = (zo + acc[nt][r] + bb_) * expf(sc);
            ldjp += sc;
            zt[w][m * 72 + d] = (half_t)zn;
        }
    }
#pragma unroll
    for (int off = 16; off > 0; off >>= 1) ldjp += __shfl_xor(ldjp, off, 32);
    if (lane == 0) atomicAdd(&ldj[b], ldjp);

    __syncthreads();

    // z <- z @ conv_W (64x64) via WMMA, write back f32
    v16h az[2];
#pragma unroll
    for (int kt = 0; kt < 2; ++kt)
        az[kt] = ldA(&zt[w][mrow * 72 + kt * 32 + hi * 16]);
#pragma unroll
    for (int nt = 0; nt < 4; ++nt) {
        v8f a = zero8();
#pragma unroll
        for (int kt = 0; kt < 2; ++kt) {
            const half_t* bp = convWt + (size_t)(nt * 16 + mrow) * 64 + kt * 32 + hi * 16;
            a = wmma16(az[kt], ldB(bp), a);
        }
#pragma unroll
        for (int r = 0; r < 8; ++r)
            zbuf[(size_t)(token0 + mr + r) * DD + nt * 16 + mrow] = a[r];
    }
}

// ---------------------------------------------------------------------------
// Stage 2 (per flow): coupling MLP. 64 tokens per block, 8 waves.
// X = [z*mask, ext] (64x128) -> gelu(XW0+b0) -> gelu(H1 W1+b1) -> H2 W2+b2.
// Only output columns 64..127 computed (mask kills d<32 entirely).
// B fragments double-buffered in registers; ext staged by async-LDS copy.
// ---------------------------------------------------------------------------
__global__ __launch_bounds__(256) void flow_stage2(
    const half_t* __restrict__ extq, float* __restrict__ zbuf,
    const half_t* __restrict__ W0t,  // [512][128]
    const float* __restrict__ b0,
    const half_t* __restrict__ W1t,  // [512][512]
    const float* __restrict__ b1,
    const half_t* __restrict__ W2t,  // [128][512]
    const float* __restrict__ b2,
    const float* __restrict__ scal,  // [64]
    float* __restrict__ ldj)
{
    __shared__ __align__(16) half_t Xs[64 * 128];
    __shared__ __align__(16) half_t H1[64 * 520];
    __shared__ __align__(16) half_t H2[64 * 520];
    const int LDH = 520;

    int tid = threadIdx.x;
    int lane = tid & 31, w = tid >> 5;
    int token0 = blockIdx.x * 64;
    int b = token0 / SS;
    int mrow = lane & 15;
    int hi = (lane >> 4) & 1;

    // ---- build X in LDS: [z(d<32), 0, ext] ----
    // ext half: async global->LDS, 16B per lane, 2 chunks per thread
#pragma unroll
    for (int i = 0; i < 2; ++i) {
        int idx = tid + i * 256;          // 512 chunks of 16B = 8KB
        int m = idx >> 3, c8 = idx & 7;   // row, 8-half chunk in ext row
        uint32_t la = (uint32_t)(uintptr_t)&Xs[m * 128 + 64 + c8 * 8];
        async_copy_b128(la, extq + (size_t)(token0 + m) * EE + c8 * 8);
    }
    // z half (masked) overlapped with the async copy: 64x64 elems, 16/thread
#pragma unroll
    for (int i = 0; i < 16; ++i) {
        int idx = tid + i * 256;
        int m = idx >> 6, c = idx & 63;
        half_t v = (c < 32) ? (half_t)zbuf[(size_t)(token0 + m) * DD + c]
                            : (half_t)0.0f;
        Xs[m * 128 + c] = v;
    }
    wait_async0();
    __syncthreads();

    // ---- Layer 0: H1 = gelu(X @ W0 + b0), wave strip = 64 cols ----
    {
        int nb = w * 64;
        v8f acc[4][4];
#pragma unroll
        for (int rt = 0; rt < 4; ++rt)
#pragma unroll
            for (int ct = 0; ct < 4; ++ct) acc[rt][ct] = zero8();

        v16h bf[4];
#pragma unroll
        for (int ct = 0; ct < 4; ++ct)
            bf[ct] = ldB(W0t + (size_t)(nb + ct * 16 + mrow) * 128 + hi * 16);
#pragma unroll
        for (int kt = 0; kt < 4; ++kt) {
            v16h bfn[4];
            if (kt < 3) {
#pragma unroll
                for (int ct = 0; ct < 4; ++ct)
                    bfn[ct] = ldB(W0t + (size_t)(nb + ct * 16 + mrow) * 128 +
                                  (kt + 1) * 32 + hi * 16);
            }
            v16h af[4];
#pragma unroll
            for (int rt = 0; rt < 4; ++rt)
                af[rt] = ldA(&Xs[(rt * 16 + mrow) * 128 + kt * 32 + hi * 16]);
#pragma unroll
            for (int ct = 0; ct < 4; ++ct)
#pragma unroll
                for (int rt = 0; rt < 4; ++rt)
                    acc[rt][ct] = wmma16(af[rt], bf[ct], acc[rt][ct]);
            if (kt < 3) {
#pragma unroll
                for (int ct = 0; ct < 4; ++ct) bf[ct] = bfn[ct];
            }
        }
#pragma unroll
        for (int ct = 0; ct < 4; ++ct) {
            int col = nb + ct * 16 + mrow;
            float bias = b0[col];
#pragma unroll
            for (int rt = 0; rt < 4; ++rt)
#pragma unroll
                for (int r = 0; r < 8; ++r) {
                    float x = acc[rt][ct][r] + bias;
                    H1[(rt * 16 + hi * 8 + r) * LDH + col] = (half_t)gelu_exact(x);
                }
        }
    }
    __syncthreads();

    // ---- Layer 1: H2 = gelu(H1 @ W1 + b1), B double-buffered ----
    {
        int nb = w * 64;
        v8f acc[4][4];
#pragma unroll
        for (int rt = 0; rt < 4; ++rt)
#pragma unroll
            for (int ct = 0; ct < 4; ++ct) acc[rt][ct] = zero8();

        v16h bf[4];
#pragma unroll
        for (int ct = 0; ct < 4; ++ct)
            bf[ct] = ldB(W1t + (size_t)(nb + ct * 16 + mrow) * 512 + hi * 16);
#pragma unroll
        for (int kt = 0; kt < 16; ++kt) {
            v16h bfn[4];
            if (kt < 15) {
#pragma unroll
                for (int ct = 0; ct < 4; ++ct)
                    bfn[ct] = ldB(W1t + (size_t)(nb + ct * 16 + mrow) * 512 +
                                  (kt + 1) * 32 + hi * 16);
            }
            v16h af[4];
#pragma unroll
            for (int rt = 0; rt < 4; ++rt)
                af[rt] = ldA(&H1[(rt * 16 + mrow) * LDH + kt * 32 + hi * 16]);
#pragma unroll
            for (int ct = 0; ct < 4; ++ct)
#pragma unroll
                for (int rt = 0; rt < 4; ++rt)
                    acc[rt][ct] = wmma16(af[rt], bf[ct], acc[rt][ct]);
            if (kt < 15) {
#pragma unroll
                for (int ct = 0; ct < 4; ++ct) bf[ct] = bfn[ct];
            }
        }
#pragma unroll
        for (int ct = 0; ct < 4; ++ct) {
            int col = nb + ct * 16 + mrow;
            float bias = b1[col];
#pragma unroll
            for (int rt = 0; rt < 4; ++rt)
#pragma unroll
                for (int r = 0; r < 8; ++r) {
                    float x = acc[rt][ct][r] + bias;
                    H2[(rt * 16 + hi * 8 + r) * LDH + col] = (half_t)gelu_exact(x);
                }
        }
    }
    __syncthreads();

    // ---- Layer 2: only cols 64..127 (d = 32..63). wave -> (rt, 32-col half)
    {
        int rt = w >> 1;
        int nb = 64 + (w & 1) * 32;
        v8f acc[2];
        acc[0] = zero8();
        acc[1] = zero8();
        v16h bf[2];
#pragma unroll
        for (int ct = 0; ct < 2; ++ct)
            bf[ct] = ldB(W2t + (size_t)(nb + ct * 16 + mrow) * 512 + hi * 16);
#pragma unroll
        for (int kt = 0; kt < 16; ++kt) {
            v16h bfn[2];
            if (kt < 15) {
#pragma unroll
                for (int ct = 0; ct < 2; ++ct)
                    bfn[ct] = ldB(W2t + (size_t)(nb + ct * 16 + mrow) * 512 +
                                  (kt + 1) * 32 + hi * 16);
            }
            v16h af = ldA(&H2[(rt * 16 + mrow) * LDH + kt * 32 + hi * 16]);
#pragma unroll
            for (int ct = 0; ct < 2; ++ct) acc[ct] = wmma16(af, bf[ct], acc[ct]);
            if (kt < 15) {
#pragma unroll
                for (int ct = 0; ct < 2; ++ct) bf[ct] = bfn[ct];
            }
        }
        float ldjp = 0.0f;
#pragma unroll
        for (int ct = 0; ct < 2; ++ct) {
            int gcol = nb + ct * 16 + mrow;  // 64..127; parity == lane parity
            float bias = b2[gcol];
            int d = gcol >> 1;               // 32..63
            float sf = expf(scal[d]);
            float denom = fmaxf(sf, 1.0f);
#pragma unroll
            for (int r = 0; r < 8; ++r) {
                float v = acc[ct][r] + bias;
                float partner = __shfl_xor(v, 1, 32);
                if ((lane & 1) == 0) {
                    float sp = tanhf(v / denom) * sf;  // v = s (even col)
                    int m = rt * 16 + hi * 8 + r;
                    size_t zi = (size_t)(token0 + m) * DD + d;
                    float zo = zbuf[zi];
                    zbuf[zi] = (zo + partner) * expf(sp);  // partner = t
                    ldjp += sp;
                }
            }
        }
#pragma unroll
        for (int off = 16; off > 0; off >>= 1) ldjp += __shfl_xor(ldjp, off, 32);
        if (lane == 0) atomicAdd(&ldj[b], ldjp);
    }
}

// ---------------------------------------------------------------------------
// Host orchestration
// ---------------------------------------------------------------------------
extern "C" void kernel_launch(void* const* d_in, const int* in_sizes, int n_in,
                              void* d_out, int out_size, void* d_ws, size_t ws_size,
                              hipStream_t stream) {
    (void)in_sizes; (void)n_in; (void)out_size; (void)ws_size;

    const float* z_in      = (const float*)d_in[0];
    const int*   categ     = (const int*)d_in[1];
    const float* embed     = (const float*)d_in[2];
    const float* actnorm_W = (const float*)d_in[3];
    const float* actnorm_b = (const float*)d_in[4];
    const float* conv_W    = (const float*)d_in[5];
    const float* scal      = (const float*)d_in[6];
    const float* W0        = (const float*)d_in[7];
    const float* b0        = (const float*)d_in[8];
    const float* W1        = (const float*)d_in[9];
    const float* b1        = (const float*)d_in[10];
    const float* W2        = (const float*)d_in[11];
    const float* b2        = (const float*)d_in[12];

    float* out_z   = (float*)d_out;
    float* out_ldj = out_z + (size_t)NTOK * DD;

    // workspace carve-up
    char* p = (char*)d_ws;
    float* zbuf = (float*)p;  p += (size_t)NTOK * DD * sizeof(float);
    half_t* extq = (half_t*)p; p += (size_t)NTOK * EE * sizeof(half_t);
    half_t *anWt[FF], *convWt[FF], *W0t[FF], *W1t[FF], *W2t[FF];
    for (int f = 0; f < FF; ++f) {
        anWt[f]   = (half_t*)p; p += (size_t)128 * 64 * sizeof(half_t);
        convWt[f] = (half_t*)p; p += (size_t)64 * 64 * sizeof(half_t);
        W0t[f]    = (half_t*)p; p += (size_t)512 * 128 * sizeof(half_t);
        W1t[f]    = (half_t*)p; p += (size_t)512 * 512 * sizeof(half_t);
        W2t[f]    = (half_t*)p; p += (size_t)128 * 512 * sizeof(half_t);
    }

    // z -> workspace (flows update in place)
    hipMemcpyAsync(zbuf, z_in, (size_t)NTOK * DD * sizeof(float),
                   hipMemcpyDeviceToDevice, stream);

    // pack weights (f16, B-operand layout)
    for (int f = 0; f < FF; ++f) {
        pack_bt<<<(64 * 128 + 255) / 256, 256, 0, stream>>>(
            actnorm_W + (size_t)f * 64 * 128, anWt[f], 64, 128);
        pack_bt<<<(64 * 64 + 255) / 256, 256, 0, stream>>>(
            conv_W + (size_t)f * 64 * 64, convWt[f], 64, 64);
        pack_bt<<<(128 * 512 + 255) / 256, 256, 0, stream>>>(
            W0 + (size_t)f * 128 * 512, W0t[f], 128, 512);
        pack_bt<<<(512 * 512 + 255) / 256, 256, 0, stream>>>(
            W1 + (size_t)f * 512 * 512, W1t[f], 512, 512);
        pack_bt<<<(512 * 128 + 255) / 256, 256, 0, stream>>>(
            W2 + (size_t)f * 512 * 128, W2t[f], 512, 128);
    }

    // ext gather
    gather_ext<<<(NTOK * EE + 255) / 256, 256, 0, stream>>>(embed, categ, extq);

    // ldj init (overwrites every call -> deterministic base before atomics)
    logdet_init<<<1, 64, 0, stream>>>(conv_W, out_ldj);

    // flows
    for (int f = 0; f < FF; ++f) {
        flow_stage1<<<NTOK / 16 / 8, 256, 0, stream>>>(
            extq, zbuf, anWt[f], actnorm_b + (size_t)f * 128, convWt[f], out_ldj);
        flow_stage2<<<NTOK / 64, 256, 0, stream>>>(
            extq, zbuf, W0t[f], b0 + (size_t)f * 512, W1t[f], b1 + (size_t)f * 512,
            W2t[f], b2 + (size_t)f * 128, scal + (size_t)f * 64, out_ldj);
    }

    // final z -> d_out
    hipMemcpyAsync(out_z, zbuf, (size_t)NTOK * DD * sizeof(float),
                   hipMemcpyDeviceToDevice, stream);
}
